// AttentionFusionBlock_18004502905389
// MI455X (gfx1250) — compile-verified, matrix operand order checked by reading, not run on hardware
//
#include <hip/hip_runtime.h>
#include <hip/hip_bf16.h>

// ---------------------------------------------------------------------------
// CDNA5 (gfx1250) fused attention block. All matmuls: v_wmma_f32_16x16x32_bf16.
// Intermediates stored bf16; Q/K/attn-out/hidden token-major [N][C] so all
// staging is contiguous 16B chunks (async-to-LDS) and epilogues are two b128
// stores per lane. V channel-major (A-fragment wants key-contiguous).
// ---------------------------------------------------------------------------

typedef __bf16 v16bf __attribute__((ext_vector_type(16)));
typedef __bf16 v8bf  __attribute__((ext_vector_type(8)));
typedef __bf16 v4bf  __attribute__((ext_vector_type(4)));
typedef float  v8f   __attribute__((ext_vector_type(8)));
typedef int    v4i   __attribute__((ext_vector_type(4)));

#define D_MODEL   256
#define N_TOK     4096
#define NUM_HEADS 8
#define HEAD_DIM  32
#define BATCH     2

#if defined(__has_builtin)
#if __has_builtin(__builtin_amdgcn_global_load_async_to_lds_b128)
#define HAVE_ASYNC_LDS 1
#endif
#endif

__device__ __forceinline__ float gelu_exact(float x) {
    return 0.5f * x * (1.0f + erff(x * 0.70710678118654752f));
}

// Copy one 16-byte bf16 chunk global -> LDS (async if toolchain exposes it).
__device__ __forceinline__ void stage16(const __bf16* g, __bf16* l) {
#ifdef HAVE_ASYNC_LDS
    __builtin_amdgcn_global_load_async_to_lds_b128(
        (__attribute__((address_space(1))) v4i*)(__bf16*)g,
        (__attribute__((address_space(3))) v4i*)l, 0, 0);
#else
    *(v8bf*)l = *(const v8bf*)g;
#endif
}

__device__ __forceinline__ void stage_wait() {
#ifdef HAVE_ASYNC_LDS
#if __has_builtin(__builtin_amdgcn_s_wait_asynccnt)
    __builtin_amdgcn_s_wait_asynccnt(0);
#else
    asm volatile("s_wait_asynccnt 0" ::: "memory");
#endif
#endif
}

// Load a 16-element bf16 fragment from two contiguous 16-byte LDS chunks.
__device__ __forceinline__ v16bf frag16(const __bf16* lo, const __bf16* hi) {
    v8bf a = *(const v8bf*)lo;
    v8bf b = *(const v8bf*)hi;
    return __builtin_shufflevector(a, b, 0, 1, 2, 3, 4, 5, 6, 7,
                                   8, 9, 10, 11, 12, 13, 14, 15);
}

__device__ __forceinline__ v4bf cvt4(float4 f) {
    v4bf r;
    r[0] = (__bf16)f.x; r[1] = (__bf16)f.y;
    r[2] = (__bf16)f.z; r[3] = (__bf16)f.w;
    return r;
}

// ---------------------------------------------------------------------------
// GEMM:  Y(M x Ntok over tokens) = Wm(M x Kdim) @ X + bias [+epi]
//   TX = float : X channel-major [Kdim][Ntok] (f32 inputs)
//   TX = __bf16: X token-major  [Ntok][Kdim] (contiguous 16B async staging)
//   OUT = 0: f32 channel-major | 1: bf16 token-major | 2: bf16 channel-major
//   RES : add f32 channel-major residual (OUT==0 only) | GELU: exact gelu
// Block tile 64x64, 8 waves, 2 WMMAs/wave per 32-deep K step.
// ---------------------------------------------------------------------------
template <typename TX, int OUT, bool RES, bool GELU>
__global__ void __launch_bounds__(256)
gemm_ch_kernel(const float* __restrict__ Wm, const TX* __restrict__ X,
               const float* __restrict__ bias, const float* __restrict__ resid,
               void* __restrict__ Yv, int M, int Kdim, int Ntok)
{
    constexpr bool XF32 = (sizeof(TX) == 4);
    const int n0 = blockIdx.x * 64;
    const int m0 = blockIdx.y * 64;
    const int bz = blockIdx.z;
    X += (size_t)bz * Kdim * Ntok;
    if (RES) resid += (size_t)bz * M * Ntok;

    __shared__ __align__(16) __bf16 Wl[64][40];    // [m][k]
    __shared__ __align__(16) __bf16 XlT[64][40];   // [n][k]

    const int tid  = threadIdx.x;
    const int lane = tid & 31;
    const int wave = tid >> 5;
    const int wm   = wave & 3;
    const int wn   = wave >> 2;
    const int arow = lane & 15;
    const int kb   = (lane >> 4) << 3;    // per-lane K-chunk base (0 or 8)
    const int ncol = lane & 15;

    v8f acc0 = {}, acc1 = {};

    for (int k0 = 0; k0 < Kdim; k0 += 32) {
        __syncthreads();
        // stage W tile (64x32 f32 -> bf16)
        for (int idx = tid * 4; idx < 64 * 32; idx += 1024) {
            int r = idx >> 5, c = idx & 31;
            float4 f = *(const float4*)&Wm[(size_t)(m0 + r) * Kdim + k0 + c];
            *(v4bf*)&Wl[r][c] = cvt4(f);
        }
        if constexpr (XF32) {
            // channel-major f32 X: coalesced float4 reads, b16 transpose scatter
            for (int idx = tid * 4; idx < 32 * 64; idx += 1024) {
                int k = idx >> 6, n = idx & 63;
                float4 f = *(const float4*)&X[(size_t)(k0 + k) * Ntok + n0 + n];
                XlT[n + 0][k] = (__bf16)f.x;
                XlT[n + 1][k] = (__bf16)f.y;
                XlT[n + 2][k] = (__bf16)f.z;
                XlT[n + 3][k] = (__bf16)f.w;
            }
            if (k0 + 32 < Kdim)
                __builtin_prefetch(
                    (const void*)&X[(size_t)(k0 + 32 + (tid >> 6)) * Ntok +
                                    n0 + (tid & 63)], 0, 1);
        } else {
            // token-major bf16 X: one contiguous 16B async chunk per thread
            int n = tid >> 2, k8 = (tid & 3) * 8;
            stage16((const __bf16*)&X[(size_t)(n0 + n) * Kdim + k0 + k8], &XlT[n][k8]);
            stage_wait();
        }
        __syncthreads();

        v16bf a  = frag16(&Wl[wm * 16 + arow][kb],       &Wl[wm * 16 + arow][kb + 16]);
        v16bf b0 = frag16(&XlT[wn * 32 + ncol][kb],      &XlT[wn * 32 + ncol][kb + 16]);
        v16bf b1 = frag16(&XlT[wn * 32 + 16 + ncol][kb], &XlT[wn * 32 + 16 + ncol][kb + 16]);

        acc0 = __builtin_amdgcn_wmma_f32_16x16x32_bf16(false, a, false, b0,
                                                       (short)0, acc0, false, false);
        acc1 = __builtin_amdgcn_wmma_f32_16x16x32_bf16(false, a, false, b1,
                                                       (short)0, acc1, false, false);
    }

    // D layout: VGPR r -> M = mb + r (mb = m0+wm*16+kb), N = n / n+16.
    const int mb = m0 + wm * 16 + kb;
    const int n  = n0 + wn * 32 + ncol;

    if constexpr (OUT == 0) {
        float* Y = (float*)Yv + (size_t)bz * M * Ntok;
#pragma unroll
        for (int r = 0; r < 8; ++r) {
            int m = mb + r;
            float v0 = acc0[r] + bias[m];
            float v1 = acc1[r] + bias[m];
            if (RES) {
                v0 += resid[(size_t)m * Ntok + n];
                v1 += resid[(size_t)m * Ntok + n + 16];
            }
            if (GELU) { v0 = gelu_exact(v0); v1 = gelu_exact(v1); }
            Y[(size_t)m * Ntok + n]      = v0;
            Y[(size_t)m * Ntok + n + 16] = v1;
        }
    } else if constexpr (OUT == 1) {
        // bf16 token-major: per lane, 8 consecutive m -> one b128 store each
        __bf16* Y = (__bf16*)Yv + (size_t)bz * M * Ntok;
        v8bf y0, y1;
#pragma unroll
        for (int r = 0; r < 8; ++r) {
            float v0 = acc0[r] + bias[mb + r];
            float v1 = acc1[r] + bias[mb + r];
            if (GELU) { v0 = gelu_exact(v0); v1 = gelu_exact(v1); }
            y0[r] = (__bf16)v0;
            y1[r] = (__bf16)v1;
        }
        *(v8bf*)&Y[(size_t)n * M + mb]        = y0;
        *(v8bf*)&Y[(size_t)(n + 16) * M + mb] = y1;
    } else {
        // bf16 channel-major (V projection)
        __bf16* Y = (__bf16*)Yv + (size_t)bz * M * Ntok;
#pragma unroll
        for (int r = 0; r < 8; ++r) {
            int m = mb + r;
            Y[(size_t)m * Ntok + n]      = (__bf16)(acc0[r] + bias[m]);
            Y[(size_t)m * Ntok + n + 16] = (__bf16)(acc1[r] + bias[m]);
        }
    }
}

// ---------------------------------------------------------------------------
// Flash attention, transposed-score formulation:
//   S^T = K_tile x Q  (C layout: lane = query) ; O^T = V_tile x P^T.
// S^T C-layout == P^T B-fragment layout, so no P transpose; softmax stats are
// lane-local scalars (15 VALU + one __shfl_xor(16) per reduction).
// Q,K token-major bf16; V channel-major bf16; out token-major bf16.
// One workgroup = 4 waves = 64 queries of one (batch, head).
// ---------------------------------------------------------------------------
__global__ void __launch_bounds__(128)
attn_kernel(const __bf16* __restrict__ QT, const __bf16* __restrict__ KT,
            const __bf16* __restrict__ Vc, __bf16* __restrict__ OT)
{
    const int qb   = blockIdx.x * 64;
    const int head = blockIdx.y;
    const int bz   = blockIdx.z;
    const __bf16* QTb = QT + (size_t)bz * N_TOK * D_MODEL + head * HEAD_DIM; // [q][c]
    const __bf16* KTb = KT + (size_t)bz * N_TOK * D_MODEL + head * HEAD_DIM; // [k][c]
    const __bf16* Vcb = Vc + (size_t)bz * D_MODEL * N_TOK
                           + (size_t)head * HEAD_DIM * N_TOK;                // [c][k]
    __bf16*       OTb = OT + (size_t)bz * N_TOK * D_MODEL + head * HEAD_DIM; // [q][c]
    const float scale = 0.17677669529663689f;   // 1/sqrt(32)

    __shared__ __align__(16) __bf16 QsT[64][40];        // [q][c]
    __shared__ __align__(16) __bf16 KsT[32][40];        // [k][c]
    __shared__ __align__(16) __bf16 Vs[HEAD_DIM][40];   // [c][k]

    const int tid  = threadIdx.x;
    const int lane = tid & 31;
    const int wave = tid >> 5;
    const int qw   = wave * 16;
    const int nc   = lane & 15;
    const int kb   = (lane >> 4) << 3;

    // stage Q block (64q x 32c): 256 contiguous 16B chunks
    for (int idx = tid; idx < 256; idx += 128) {
        int q = idx >> 2, c8 = (idx & 3) * 8;
        stage16(&QTb[(size_t)(qb + q) * D_MODEL + c8], &QsT[q][c8]);
    }
    stage_wait();
    __syncthreads();

    // B fragment for Q (contiguous halves from token-major rows)
    v16bf bq = frag16(&QsT[qw + nc][kb], &QsT[qw + nc][kb + 16]);

    v8f   oa0 = {}, oa1 = {};          // O^T: c tiles [0,16) and [16,32)
    float mrow = -INFINITY, lrow = 0.0f;

    const int sk = tid >> 2;           // staged row (key for K, channel for V)
    const int s8 = (tid & 3) * 8;      // 16B chunk within row

    for (int kblk = 0; kblk < N_TOK; kblk += 32) {
        __syncthreads();
        stage16(&KTb[(size_t)(kblk + sk) * D_MODEL + s8], &KsT[sk][s8]);
        stage16(&Vcb[(size_t)sk * N_TOK + kblk + s8],     &Vs[sk][s8]);
        stage_wait();
        __syncthreads();

        // A fragments for K tiles: A[m=key, k=c] contiguous
        v16bf ak0 = frag16(&KsT[nc][kb],      &KsT[nc][kb + 16]);
        v16bf ak1 = frag16(&KsT[16 + nc][kb], &KsT[16 + nc][kb + 16]);

        v8f z = {};
        v8f s0 = __builtin_amdgcn_wmma_f32_16x16x32_bf16(false, ak0, false, bq,
                                                         (short)0, z, false, false);
        v8f s1 = __builtin_amdgcn_wmma_f32_16x16x32_bf16(false, ak1, false, bq,
                                                         (short)0, z, false, false);

        // Lane-local online softmax (this lane's 16 scores share query nc)
        float mx = -INFINITY;
#pragma unroll
        for (int r = 0; r < 8; ++r)
            mx = fmaxf(mx, fmaxf(s0[r] * scale, s1[r] * scale));
        mx = fmaxf(mx, __shfl_xor(mx, 16, 32));
        float mn = fmaxf(mrow, mx);

        float p0[8], p1[8];
        float rs = 0.0f;
#pragma unroll
        for (int r = 0; r < 8; ++r) {
            p0[r] = __expf(s0[r] * scale - mn);
            p1[r] = __expf(s1[r] * scale - mn);
            rs += p0[r] + p1[r];
        }
        rs += __shfl_xor(rs, 16, 32);

        float al = __expf(mrow - mn);
        lrow = lrow * al + rs;
        mrow = mn;
#pragma unroll
        for (int r = 0; r < 8; ++r) { oa0[r] *= al; oa1[r] *= al; }

        // P^T B-fragment directly from score registers (layouts coincide)
        v16bf pv;
#pragma unroll
        for (int r = 0; r < 8; ++r) {
            pv[r]     = (__bf16)p0[r];
            pv[8 + r] = (__bf16)p1[r];
        }

        // A fragments for V tiles: A[m=c, k=key] contiguous
        v16bf av0 = frag16(&Vs[nc][kb],      &Vs[nc][kb + 16]);
        v16bf av1 = frag16(&Vs[16 + nc][kb], &Vs[16 + nc][kb + 16]);

        oa0 = __builtin_amdgcn_wmma_f32_16x16x32_bf16(false, av0, false, pv,
                                                      (short)0, oa0, false, false);
        oa1 = __builtin_amdgcn_wmma_f32_16x16x32_bf16(false, av1, false, pv,
                                                      (short)0, oa1, false, false);
    }

    // O^T layout: VGPR r -> c = r + kb (+16 for tile1), q = qw + nc.
    // Token-major epilogue: 8 consecutive c -> two b128 stores per lane.
    float inv = 1.0f / lrow;
    int   q   = qb + qw + nc;
    v8bf  y0, y1;
#pragma unroll
    for (int r = 0; r < 8; ++r) {
        y0[r] = (__bf16)(oa0[r] * inv);
        y1[r] = (__bf16)(oa1[r] * inv);
    }
    __bf16* orow = OTb + (size_t)q * D_MODEL;
    *(v8bf*)&orow[kb]      = y0;
    *(v8bf*)&orow[16 + kb] = y1;
}

// ---------------------------------------------------------------------------
extern "C" void kernel_launch(void* const* d_in, const int* in_sizes, int n_in,
                              void* d_out, int out_size, void* d_ws, size_t ws_size,
                              hipStream_t stream)
{
    const float* q_feat = (const float*)d_in[0];   // (B,C,N) channel-major f32
    const float* l_feat = (const float*)d_in[1];
    const float* Wq = (const float*)d_in[2];
    const float* bq = (const float*)d_in[3];
    const float* Wk = (const float*)d_in[4];
    const float* bk = (const float*)d_in[5];
    const float* Wv = (const float*)d_in[6];
    const float* bv = (const float*)d_in[7];
    const float* Wo = (const float*)d_in[8];
    const float* bo = (const float*)d_in[9];
    const float* W1 = (const float*)d_in[10];
    const float* b1 = (const float*)d_in[11];
    const float* W2 = (const float*)d_in[12];
    const float* b2 = (const float*)d_in[13];

    const int C = D_MODEL, C4 = 4 * D_MODEL, NT = N_TOK, B = BATCH;
    const size_t BCN = (size_t)B * C * NT;

    __bf16* QT16  = (__bf16*)d_ws;     // [B][N][C]  token-major
    __bf16* KT16  = QT16 + BCN;        // [B][N][C]  token-major
    __bf16* Vc16  = KT16 + BCN;        // [B][C][N]  channel-major
    __bf16* AOT16 = Vc16 + BCN;        // [B][N][C]  token-major
    __bf16* HdT16 = AOT16 + BCN;       // [B][N][4C] token-major

    float* out = (float*)d_out;        // `out` (attention + residual), f32 ch-major

    dim3 blk(256);
    // Q/K/V projections (f32 in, bf16 out)
    gemm_ch_kernel<float, 1, false, false><<<dim3(NT / 64, C / 64, B), blk, 0, stream>>>(
        Wq, q_feat, bq, nullptr, QT16, C, C, NT);
    gemm_ch_kernel<float, 1, false, false><<<dim3(NT / 64, C / 64, B), blk, 0, stream>>>(
        Wk, l_feat, bk, nullptr, KT16, C, C, NT);
    gemm_ch_kernel<float, 2, false, false><<<dim3(NT / 64, C / 64, B), blk, 0, stream>>>(
        Wv, l_feat, bv, nullptr, Vc16, C, C, NT);
    // attention (bf16 in/out)
    attn_kernel<<<dim3(NT / 64, NUM_HEADS, B), dim3(128), 0, stream>>>(
        QT16, KT16, Vc16, AOT16);
    // output projection (bf16 X) + residual(query_feat) -> out (f32, d_out)
    gemm_ch_kernel<__bf16, 0, true, false><<<dim3(NT / 64, C / 64, B), blk, 0, stream>>>(
        Wo, AOT16, bo, q_feat, out, C, C, NT);
    // MLP fc1 (f32 X) + GELU -> bf16 token-major hidden
    gemm_ch_kernel<float, 1, false, true><<<dim3(NT / 64, C4 / 64, B), blk, 0, stream>>>(
        W1, out, b1, nullptr, HdT16, C4, C, NT);
    // MLP fc2 (bf16 X) + residual(out) -> final (d_out; element read only by writer)
    gemm_ch_kernel<__bf16, 0, true, false><<<dim3(NT / 64, C / 64, B), blk, 0, stream>>>(
        W2, HdT16, b2, out, out, C, C4, NT);
}
